// Mamba_67405216743447
// MI455X (gfx1250) — compile-verified, matrix operand order checked by reading, not run on hardware
//
#include <hip/hip_runtime.h>
#include <hip/hip_bf16.h>
#include <math.h>
#include <stdint.h>

// ---------------------------------------------------------------------------
// Mamba forward for MI455X (gfx1250, wave32, WMMA).
//   Projections: f16-operand / f32-accumulate WMMA GEMMs
//   (v_wmma_f32_16x16x32_f16). Tiles are moved with the CDNA5 async
//   global->LDS datapath (global_load_async_to_lds_b128 / ASYNCcnt) in a
//   double-buffered pipeline, so WMMAs overlap the DMA of the next K tile.
//   Scan: 64 states/channel in VGPRs, B_t/C_t double-buffered through LDS.
// ---------------------------------------------------------------------------

typedef __attribute__((ext_vector_type(16))) _Float16 v16h;
typedef __attribute__((ext_vector_type(8)))  _Float16 v8h;
typedef __attribute__((ext_vector_type(4)))  _Float16 v4h;
typedef __attribute__((ext_vector_type(8)))  float    v8f;

#define B_SZ    2
#define L_SEQ   4096
#define D_MODEL 1024
#define D_INNER 2048
#define D_STATE 64
#define DT_RANK 128
#define BL      (B_SZ * L_SEQ)          // 8192 token rows

#define TM 128
#define TN 64
#define TK 32
#define LDSP 40   // padded LDS row stride (halves): 80 B = 5 x 16 B chunks

__device__ __forceinline__ float softplusf(float x) {
  return (x > 20.0f) ? x : log1pf(__expf(x));
}
__device__ __forceinline__ float siluf(float x) {
  return x / (1.0f + __expf(-x));
}

__device__ __forceinline__ void async_copy_b128(uint32_t lds_off, const void* g) {
  asm volatile("global_load_async_to_lds_b128 %0, %1, off"
               :: "v"(lds_off), "v"((uint64_t)(uintptr_t)g) : "memory");
}
__device__ __forceinline__ void wait_async0() {
  asm volatile("s_wait_asynccnt 0x0" ::: "memory");
}

// ---------------- f32 -> f16 conversion (vectorized, n % 4 == 0) -----------
__global__ __launch_bounds__(256) void cvt_f32_f16(
    const float* __restrict__ s, _Float16* __restrict__ d, size_t n)
{
  const size_t i = ((size_t)blockIdx.x * 256 + threadIdx.x) * 4;
  if (i + 3 < n) {
    const float4 v = *(const float4*)(s + i);
    v4h o = { (_Float16)v.x, (_Float16)v.y, (_Float16)v.z, (_Float16)v.w };
    *(v4h*)(d + i) = o;
  }
}

// ---------------- WMMA GEMM:  Out[M,N] = A[M,K](lda) * W[N,K]^T ------------
// MODE 0: f32 store.  MODE 1: softplus(v + bias[n]).  MODE 2: f32 + f16 copy.
template<int MODE>
__global__ __launch_bounds__(256) void gemm_f16wmma(
    const _Float16* __restrict__ A, int lda,       // A: M x K (f16)
    const _Float16* __restrict__ W,                // W: N x K row-major (f16)
    float* __restrict__ Out, int ldo,              // Out: M x N (f32)
    _Float16* __restrict__ Out16,                  // MODE 2 mirror (ld = ldo)
    const float* __restrict__ bias,
    int M, int N, int K)
{
  __shared__ _Float16 As[2][TM][LDSP];  // 2 x 10240 B
  __shared__ _Float16 Bs[2][TN][LDSP];  // 2 x  5120 B

  const int tid  = threadIdx.x;
  const int wave = tid >> 5;
  const int lane = tid & 31;
  const int m0 = blockIdx.x * TM;
  const int n0 = blockIdx.y * TN;
  const int wm = (wave & 3) * 32;        // 4 waves tile M
  const int wn = (wave >> 2) * 32;       // 2 waves tile N
  const int lrow = lane & 15;
  const int kb   = (lane < 16) ? 0 : 8;  // ISA 16-bit A/B fragment K split

  // Staging map: 16B chunks. A tile = 128 rows x 4 chunks (2 chunks/thread,
  // rows r and r+64, same K segment). B tile = 64 rows x 4 chunks (1/thread).
  const int r0 = tid >> 2;               // 0..63
  const int s0 = (tid & 3) * 8;          // K segment within tile
  const _Float16* Ap = A + (size_t)(m0 + r0) * lda + s0;
  const _Float16* Aq = A + (size_t)(m0 + r0 + 64) * lda + s0;
  const _Float16* Bp = W + (size_t)(n0 + r0) * K + s0;

  // LDS byte offsets of this thread's destination chunks (buffer 0); the
  // generic-pointer low 32 bits are the LDS offset (flat-aperture rule).
  const uint32_t aOff0 = (uint32_t)(uintptr_t)&As[0][r0][s0];
  const uint32_t aOff1 = (uint32_t)(uintptr_t)&As[0][r0 + 64][s0];
  const uint32_t bOff  = (uint32_t)(uintptr_t)&Bs[0][r0][s0];
  const uint32_t aStride = (uint32_t)sizeof(_Float16) * TM * LDSP;
  const uint32_t bStride = (uint32_t)sizeof(_Float16) * TN * LDSP;

  v8f acc[2][2] = {};

  // prologue: DMA tile 0 into buffer 0
  async_copy_b128(aOff0, Ap);
  async_copy_b128(aOff1, Aq);
  async_copy_b128(bOff,  Bp);
  wait_async0();
  __syncthreads();

  int cur = 0;
  for (int k0 = 0; k0 < K; k0 += TK) {
    if (k0 + TK < K) {                   // DMA next tile into other buffer
      const int nxt = cur ^ 1;
      async_copy_b128(aOff0 + nxt * aStride, Ap + k0 + TK);
      async_copy_b128(aOff1 + nxt * aStride, Aq + k0 + TK);
      async_copy_b128(bOff  + nxt * bStride, Bp + k0 + TK);
    }

    // Fragment gather: lane holds K in [kb,kb+8) u [16+kb,16+kb+8)
    v16h af[2], bf[2];
    #pragma unroll
    for (int i = 0; i < 2; ++i) {
      v8h lo = *(const v8h*)&As[cur][wm + i * 16 + lrow][kb];
      v8h hi = *(const v8h*)&As[cur][wm + i * 16 + lrow][16 + kb];
      af[i] = __builtin_shufflevector(lo, hi, 0,1,2,3,4,5,6,7,8,9,10,11,12,13,14,15);
    }
    #pragma unroll
    for (int j = 0; j < 2; ++j) {
      v8h lo = *(const v8h*)&Bs[cur][wn + j * 16 + lrow][kb];
      v8h hi = *(const v8h*)&Bs[cur][wn + j * 16 + lrow][16 + kb];
      bf[j] = __builtin_shufflevector(lo, hi, 0,1,2,3,4,5,6,7,8,9,10,11,12,13,14,15);
    }
    #pragma unroll
    for (int i = 0; i < 2; ++i)
      #pragma unroll
      for (int j = 0; j < 2; ++j)
        acc[i][j] = __builtin_amdgcn_wmma_f32_16x16x32_f16(
            false, af[i], false, bf[j], (short)0, acc[i][j], false, false);

    wait_async0();      // my next-tile DMA landed
    __syncthreads();    // everyone done reading cur + everyone's DMA landed
    cur ^= 1;
  }

  // C/D layout: VGPR r -> M = r (lanes 0-15) / 8+r (lanes 16-31); N = lane&15
  const int mb = m0 + wm + ((lane < 16) ? 0 : 8);
  #pragma unroll
  for (int i = 0; i < 2; ++i) {
    #pragma unroll
    for (int j = 0; j < 2; ++j) {
      const int n = n0 + wn + j * 16 + lrow;
      #pragma unroll
      for (int r = 0; r < 8; ++r) {
        const int m = mb + i * 16 + r;
        float v = acc[i][j][r];
        if (MODE == 1) v = softplusf(v + bias[n]);
        Out[(size_t)m * ldo + n] = v;
        if (MODE == 2) Out16[(size_t)m * ldo + n] = (_Float16)v;
      }
    }
  }
}

// ---------------- causal depthwise conv (width 4) + SiLU -------------------
__global__ __launch_bounds__(256) void conv_silu_kernel(
    const float* __restrict__ xz, const float* __restrict__ cw,
    const float* __restrict__ cb, float* __restrict__ xc,
    _Float16* __restrict__ xch)
{
  const size_t idx = (size_t)blockIdx.x * 256 + threadIdx.x; // BL*D_INNER
  const int d  = (int)(idx & (D_INNER - 1));
  const int bl = (int)(idx >> 11);            // D_INNER = 2^11
  const int l  = bl & (L_SEQ - 1);
  float s = cb[d];
  #pragma unroll
  for (int j = 0; j < 4; ++j) {
    const int ls = l - 3 + j;
    if (ls >= 0)
      s = fmaf(cw[d * 4 + j], xz[((size_t)(bl - l + ls)) * (2 * D_INNER) + d], s);
  }
  const float r = s * (1.0f / (1.0f + __expf(-s)));
  xc[idx]  = r;
  xch[idx] = (_Float16)r;
}

// ---------------- selective scan + gating epilogue -------------------------
__global__ __launch_bounds__(256) void scan_kernel(
    const float* __restrict__ xc,     // (BL, D_INNER) conv+silu (f32)
    const float* __restrict__ dtbuf,  // (BL, D_INNER) softplus(dt)
    const float* __restrict__ xdbl,   // (BL, 256): [128:192)=B [192:256)=C
    const float* __restrict__ xz,     // (BL, 2*D_INNER): z at [D_INNER, 2*D_INNER)
    const float* __restrict__ A_log,  // (D_INNER, D_STATE)
    const float* __restrict__ Dvec,   // (D_INNER)
    _Float16* __restrict__ y)         // (BL, D_INNER) f16 (feeds out GEMM)
{
  __shared__ float BC[2][2 * D_STATE];
  const int tid = threadIdx.x;
  const int b   = blockIdx.x >> 3;
  const int d   = ((blockIdx.x & 7) << 8) + tid;

  float Arow[D_STATE], h[D_STATE];
  #pragma unroll
  for (int n = 0; n < D_STATE; ++n) {
    Arow[n] = -__expf(A_log[(size_t)d * D_STATE + n]);
    h[n] = 0.0f;
  }
  const float Dd = Dvec[d];
  const size_t base = (size_t)b * L_SEQ;

  if (tid < 2 * D_STATE)
    BC[0][tid] = xdbl[base * 256 + DT_RANK + tid];
  __syncthreads();

  int buf = 0;
  for (int t = 0; t < L_SEQ; ++t) {
    if (t + 1 < L_SEQ && tid < 2 * D_STATE)
      BC[buf ^ 1][tid] = xdbl[(base + t + 1) * 256 + DT_RANK + tid];

    const size_t row = base + t;
    const float dtv = dtbuf[row * D_INNER + d];
    const float xv  = xc[row * D_INNER + d];
    const float zv  = xz[row * (2 * D_INNER) + D_INNER + d];
    const float dx  = dtv * xv;
    const float* Bt = &BC[buf][0];
    const float* Ct = &BC[buf][D_STATE];

    float acc = 0.0f;
    #pragma unroll
    for (int n = 0; n < D_STATE; ++n) {
      const float dA = __expf(dtv * Arow[n]);
      h[n] = fmaf(h[n], dA, dx * Bt[n]);
      acc  = fmaf(h[n], Ct[n], acc);
    }
    y[row * D_INNER + d] = (_Float16)((acc + Dd * xv) * siluf(zv));
    buf ^= 1;
    __syncthreads();
  }
}

// ---------------------------------------------------------------------------
extern "C" void kernel_launch(void* const* d_in, const int* in_sizes, int n_in,
                              void* d_out, int out_size, void* d_ws, size_t ws_size,
                              hipStream_t stream) {
  const float* hidden     = (const float*)d_in[0];
  const float* in_proj_w  = (const float*)d_in[1];
  const float* conv_w     = (const float*)d_in[2];
  const float* conv_b     = (const float*)d_in[3];
  const float* x_proj_w   = (const float*)d_in[4];
  const float* dt_proj_w  = (const float*)d_in[5];
  const float* dt_proj_b  = (const float*)d_in[6];
  const float* A_log      = (const float*)d_in[7];
  const float* Dv         = (const float*)d_in[8];
  const float* out_proj_w = (const float*)d_in[9];
  float* out = (float*)d_out;

  // ---- workspace layout: f32 region then f16 region (~379 MB total) ----
  float* xz   = (float*)d_ws;                         // BL x 4096
  float* xc   = xz   + (size_t)BL * (2 * D_INNER);    // BL x 2048
  float* xdbl = xc   + (size_t)BL * D_INNER;          // BL x 256
  float* dt   = xdbl + (size_t)BL * 256;              // BL x 2048

  _Float16* hidden_h = (_Float16*)(dt + (size_t)BL * D_INNER);
  _Float16* w_in_h   = hidden_h + (size_t)BL * D_MODEL;           // 4096x1024
  _Float16* w_x_h    = w_in_h   + (size_t)(2 * D_INNER) * D_MODEL;// 256x2048
  _Float16* w_dt_h   = w_x_h    + (size_t)256 * D_INNER;          // 2048x128
  _Float16* w_out_h  = w_dt_h   + (size_t)D_INNER * DT_RANK;      // 1024x2048
  _Float16* xc_h     = w_out_h  + (size_t)D_MODEL * D_INNER;      // BL x 2048
  _Float16* xdbl_h   = xc_h     + (size_t)BL * D_INNER;           // BL x 256
  _Float16* y_h      = xdbl_h   + (size_t)BL * 256;               // BL x 2048

  const dim3 blk(256);
  auto cvt = [&](const float* s, _Float16* dh, size_t n) {
    cvt_f32_f16<<<(unsigned)((n / 4 + 255) / 256), blk, 0, stream>>>(s, dh, n);
  };

  // 0) one-shot f16 conversions of GEMM operands
  cvt(hidden,     hidden_h, (size_t)BL * D_MODEL);
  cvt(in_proj_w,  w_in_h,   (size_t)(2 * D_INNER) * D_MODEL);
  cvt(x_proj_w,   w_x_h,    (size_t)256 * D_INNER);
  cvt(dt_proj_w,  w_dt_h,   (size_t)D_INNER * DT_RANK);
  cvt(out_proj_w, w_out_h,  (size_t)D_MODEL * D_INNER);

  // 1) xz = hidden @ in_proj_w^T            (8192 x 1024 -> 8192 x 4096)
  gemm_f16wmma<0><<<dim3(BL / TM, (2 * D_INNER) / TN), blk, 0, stream>>>(
      hidden_h, D_MODEL, w_in_h, xz, 2 * D_INNER, nullptr, nullptr,
      BL, 2 * D_INNER, D_MODEL);

  // 2) causal depthwise conv + silu on x half (writes f32 + f16)
  conv_silu_kernel<<<((size_t)BL * D_INNER) / 256, blk, 0, stream>>>(
      xz, conv_w, conv_b, xc, xc_h);

  // 3) x_dbl = xc @ x_proj_w^T  (f32 for scan + f16 mirror for dt GEMM)
  gemm_f16wmma<2><<<dim3(BL / TM, 256 / TN), blk, 0, stream>>>(
      xc_h, D_INNER, w_x_h, xdbl, 256, xdbl_h, nullptr, BL, 256, D_INNER);

  // 4) dt = softplus(x_dbl[:, :128] @ dt_proj_w^T + b)   (K=128, lda=256)
  gemm_f16wmma<1><<<dim3(BL / TM, D_INNER / TN), blk, 0, stream>>>(
      xdbl_h, 256, w_dt_h, dt, D_INNER, nullptr, dt_proj_b,
      BL, D_INNER, DT_RANK);

  // 5) selective scan + D*x + silu(z) gating -> y (f16)
  scan_kernel<<<B_SZ * (D_INNER / 256), blk, 0, stream>>>(
      xc, dt, xdbl, xz, A_log, Dv, y_h);

  // 6) out = y @ out_proj_w^T               (8192 x 2048 -> 8192 x 1024)
  gemm_f16wmma<0><<<dim3(BL / TM, D_MODEL / TN), blk, 0, stream>>>(
      y_h, D_INNER, w_out_h, out, D_MODEL, nullptr, nullptr,
      BL, D_MODEL, D_INNER);
}
